// GraphConvolution_20452634263994
// MI455X (gfx1250) — matrix operand order repaired and checked by reading, not compiled, not run on hardware
//
#include <hip/hip_runtime.h>
#include <hip/hip_fp16.h>

// GCN layer: out = relu(segment_sum(( (v@W)[edge_src] * edge_w ), edge_dst))
// Phase 1: convert v -> f16 (M padded to 10016), W -> f16 transposed (N-major)
// Phase 2: software-pipelined WMMA f16 GEMM (f32 accumulate) -> support f32
// Phase 3: zero out, edge scatter with hardware f32 atomics, relu.

typedef __attribute__((ext_vector_type(16))) _Float16 v16h;
typedef __attribute__((ext_vector_type(8)))  _Float16 v8h;
typedef __attribute__((ext_vector_type(8)))  float    v8f;

#define N_NODES 10000
#define N_EDGES 160000
#define D_IN    1024
#define D_OUT   512
#define M_TILES 626            // ceil(10000/16) + 1
#define M_PAD   (M_TILES * 16) // 10016

// ---------------- Phase 1: precision conversion ----------------

__global__ void k_convert_v(const float* __restrict__ v, _Float16* __restrict__ vh) {
    size_t i = (size_t)blockIdx.x * blockDim.x + threadIdx.x;
    if (i >= (size_t)M_PAD * D_IN) return;
    size_t row = i / D_IN;
    vh[i] = (row < N_NODES) ? (_Float16)v[i] : (_Float16)0.0f;
}

// Wt[n, k] = (f16) W[k, n]  -- N-major so B-fragment loads are K-contiguous
__global__ void k_transpose_w(const float* __restrict__ W, _Float16* __restrict__ Wt) {
    int i = blockIdx.x * blockDim.x + threadIdx.x;  // output index n*D_IN + k
    if (i >= D_IN * D_OUT) return;
    int n = i / D_IN;
    int k = i - n * D_IN;
    Wt[i] = (_Float16)W[(size_t)k * D_OUT + n];
}

// ---------------- Phase 2: WMMA GEMM (software pipelined) ----------------
// Block = 256 threads = 8 waves; wave w computes M-tile (blockIdx.x*8 + w),
// N columns [blockIdx.y*64, +64) as 4 WMMA subtiles. Next K-step's fragments
// are loaded before the current K-step's WMMAs so loads overlap matrix math.

__device__ __forceinline__ v16h load_afrag(const _Float16* __restrict__ arow, int ka) {
    union { v16h v; v8h h[2]; } af;
    af.h[0] = *reinterpret_cast<const v8h*>(arow + ka);       // K lo-quad
    af.h[1] = *reinterpret_cast<const v8h*>(arow + ka + 8);   // K next 8 (contiguous)
    return af.v;
}

__global__ void __launch_bounds__(256)
k_gemm(const _Float16* __restrict__ A,   // [M_PAD, D_IN] row-major f16
       const _Float16* __restrict__ Bt,  // [D_OUT, D_IN] N-major f16
       float* __restrict__ C) {          // [M_PAD, D_OUT] f32
    const int lane   = threadIdx.x & 31;
    const int wave   = threadIdx.x >> 5;
    const int mtile  = blockIdx.x * 8 + wave;
    if (mtile >= M_TILES) return;        // whole wave exits; EXEC stays all-1s for WMMA
    const int n0     = blockIdx.y * 64;
    const int hiHalf = (lane >> 4) & 1;  // lanes 16..31 hold the "high" K half
    const int mrow   = mtile * 16 + (lane & 15);

    const _Float16* __restrict__ arow = A  + (size_t)mrow * D_IN;
    const _Float16* __restrict__ bcol = Bt + (size_t)(n0 + (lane & 15)) * D_IN;

    v8f acc0 = {}, acc1 = {}, acc2 = {}, acc3 = {};

    // Prologue: fragments for k0 = 0
    v16h a  = load_afrag(arow, hiHalf * 16);
    const _Float16* bb = bcol + hiHalf * 16;
    v16h b0 = *reinterpret_cast<const v16h*>(bb);
    v16h b1 = *reinterpret_cast<const v16h*>(bb + 16 * D_IN);
    v16h b2 = *reinterpret_cast<const v16h*>(bb + 32 * D_IN);
    v16h b3 = *reinterpret_cast<const v16h*>(bb + 48 * D_IN);

    for (int k0 = 0; k0 < D_IN; k0 += 32) {
        const int k1 = k0 + 32;
        const bool more = (k1 < D_IN);
        v16h na = {}, nb0 = {}, nb1 = {}, nb2 = {}, nb3 = {};
        if (more) {
            // Issue next K-step's loads BEFORE this step's WMMAs.
            na = load_afrag(arow, k1 + hiHalf * 16);
            const _Float16* nb = bcol + k1 + hiHalf * 16;
            nb0 = *reinterpret_cast<const v16h*>(nb);
            nb1 = *reinterpret_cast<const v16h*>(nb + 16 * D_IN);
            nb2 = *reinterpret_cast<const v16h*>(nb + 32 * D_IN);
            nb3 = *reinterpret_cast<const v16h*>(nb + 48 * D_IN);
            // prefetch the K-step after that into WGP-level caches
            __builtin_prefetch(arow + k1 + 32, 0, 3);
        }

        acc0 = __builtin_amdgcn_wmma_f32_16x16x32_f16(false, a, false, b0, (short)0, acc0, false, false);
        acc1 = __builtin_amdgcn_wmma_f32_16x16x32_f16(false, a, false, b1, (short)0, acc1, false, false);
        acc2 = __builtin_amdgcn_wmma_f32_16x16x32_f16(false, a, false, b2, (short)0, acc2, false, false);
        acc3 = __builtin_amdgcn_wmma_f32_16x16x32_f16(false, a, false, b3, (short)0, acc3, false, false);

        if (!more) break;
        a = na; b0 = nb0; b1 = nb1; b2 = nb2; b3 = nb3;
    }

    // C/D layout: VGPR j holds row (j + hiHalf*8), column = lane&15
    const int crow0 = mtile * 16 + hiHalf * 8;
    float* cp = C + (size_t)crow0 * D_OUT + n0 + (lane & 15);
#pragma unroll
    for (int j = 0; j < 8; ++j) {
        cp[(size_t)j * D_OUT + 0]  = acc0[j];
        cp[(size_t)j * D_OUT + 16] = acc1[j];
        cp[(size_t)j * D_OUT + 32] = acc2[j];
        cp[(size_t)j * D_OUT + 48] = acc3[j];
    }
}

// ---------------- Phase 3: SpMM scatter + ReLU ----------------

__global__ void k_zero(float* __restrict__ out, int n) {
    int i = blockIdx.x * blockDim.x + threadIdx.x;
    if (i < n) out[i] = 0.0f;
}

__global__ void k_scatter(const float* __restrict__ support,
                          const int* __restrict__ esrc,
                          const int* __restrict__ edst,
                          const float* __restrict__ ew,
                          float* __restrict__ out) {
    const int e = blockIdx.x;
    const int s = esrc[e];
    const int d = edst[e];
    const float w = ew[e];
    const float* sp = support + (size_t)s * D_OUT;
    float* op = out + (size_t)d * D_OUT;
    for (int f = threadIdx.x; f < D_OUT; f += blockDim.x) {
        unsafeAtomicAdd(op + f, sp[f] * w);   // hardware global_atomic_add_f32
    }
}

__global__ void k_relu(float* __restrict__ out, int n) {
    int i = blockIdx.x * blockDim.x + threadIdx.x;
    if (i < n) out[i] = fmaxf(out[i], 0.0f);
}

// ---------------- launcher ----------------

extern "C" void kernel_launch(void* const* d_in, const int* in_sizes, int n_in,
                              void* d_out, int out_size, void* d_ws, size_t ws_size,
                              hipStream_t stream) {
    const float* v    = (const float*)d_in[0];
    const float* W    = (const float*)d_in[1];
    const int*   esrc = (const int*)d_in[2];
    const int*   edst = (const int*)d_in[3];
    const float* ew   = (const float*)d_in[4];
    float* out = (float*)d_out;

    char* ws = (char*)d_ws;
    _Float16* vh = (_Float16*)ws;                                   // 20.5 MB
    size_t off = (size_t)M_PAD * D_IN * sizeof(_Float16);
    _Float16* Wt = (_Float16*)(ws + off);                           // 1 MB
    off += (size_t)D_OUT * D_IN * sizeof(_Float16);
    float* support = (float*)(ws + off);                            // 20.5 MB

    k_convert_v<<<(M_PAD * D_IN + 255) / 256, 256, 0, stream>>>(v, vh);
    k_transpose_w<<<(D_IN * D_OUT + 255) / 256, 256, 0, stream>>>(W, Wt);

    dim3 gg((M_TILES + 7) / 8, D_OUT / 64);
    k_gemm<<<gg, 256, 0, stream>>>(vh, Wt, support);

    const int nout = N_NODES * D_OUT;
    k_zero<<<(nout + 255) / 256, 256, 0, stream>>>(out, nout);
    k_scatter<<<N_EDGES, 256, 0, stream>>>(support, esrc, edst, ew, out);
    k_relu<<<(nout + 255) / 256, 256, 0, stream>>>(out, nout);
}
